// MLP_20469814133463
// MI455X (gfx1250) — matrix operand order repaired and checked by reading, not compiled, not run on hardware
//
#include <hip/hip_runtime.h>

// ---------------------------------------------------------------------------
// Spiking MLP forward: fc1 -> LN -> LIF -> fc2 -> LN -> LIF
// Shapes: x[T=4,B=8,L=512,D=768]; W1[768,3072]; W2[3072,768]
// Flattened rows M = T*B*L = 16384, row index = t*4096 + bl.
// GEMMs in bf16 WMMA (v_wmma_f32_16x16x32_bf16), fp32 accumulate.
// ---------------------------------------------------------------------------

typedef __attribute__((ext_vector_type(16))) __bf16 v16bf;
typedef __attribute__((ext_vector_type(8)))  __bf16 v8bf;
typedef __attribute__((ext_vector_type(8)))  float  v8f;

static constexpr int M_ROWS = 16384;   // T*B*L
static constexpr int D_DIM  = 768;
static constexpr int H_DIM  = 3072;
static constexpr int BL     = 4096;    // B*L
static constexpr int T_STEPS = 4;

__device__ __forceinline__ unsigned short f2bf(float f) {
    unsigned int u = __builtin_bit_cast(unsigned int, f);
    unsigned int r = u + 0x7FFFu + ((u >> 16) & 1u);   // round-to-nearest-even
    return (unsigned short)(r >> 16);
}

// --------------------------- conversion kernels ----------------------------

__global__ __launch_bounds__(256) void f32_to_bf16_kernel(
    const float* __restrict__ in, unsigned short* __restrict__ out, int n) {
    int idx = blockIdx.x * 256 + threadIdx.x;
    if (idx < n) out[idx] = f2bf(in[idx]);
}

// in: [K, N] row-major fp32   ->   out: [N, K] bf16 (column of B contiguous)
__global__ __launch_bounds__(256) void transpose_to_bf16_kernel(
    const float* __restrict__ in, unsigned short* __restrict__ out, int K, int N) {
    int n = blockIdx.x * 256 + threadIdx.x;   // N divisible by 256
    int k = blockIdx.y;
    out[(size_t)n * K + k] = f2bf(in[(size_t)k * N + n]);
}

// ------------------------------ WMMA GEMM ----------------------------------
// C[M,N] = A[M,K](bf16) * B[K,N] + bias, with Bt stored as [N,K] bf16.
// Block: 256 threads = 8 waves. Block tile 128x128.
// Wave grid: 4 (M) x 2 (N); wave tile 32x64 = 2x4 WMMA 16x16 tiles.

__global__ __launch_bounds__(256) void wmma_gemm_bf16_kernel(
    const unsigned short* __restrict__ A,   // [M,K] bf16 row-major
    const unsigned short* __restrict__ Bt,  // [N,K] bf16
    const float* __restrict__ bias,         // [N]
    float* __restrict__ C,                  // [M,N] fp32
    int M, int N, int K) {

    const int tid   = threadIdx.x;
    const int lane  = tid & 31;
    const int wave  = tid >> 5;
    const int waveM = wave & 3;           // 0..3 -> 32 rows each
    const int waveN = wave >> 2;          // 0..1 -> 64 cols each
    const int rowBase = blockIdx.y * 128 + waveM * 32;
    const int colBase = blockIdx.x * 128 + waveN * 64;
    const int lm = lane & 15;
    const int kg = lane >> 4;             // K-group select (ISA 16-bit layouts)

    v8f acc[2][4];
#pragma unroll
    for (int tm = 0; tm < 2; ++tm)
#pragma unroll
        for (int tn = 0; tn < 4; ++tn)
#pragma unroll
            for (int r = 0; r < 8; ++r) acc[tm][tn][r] = 0.0f;

    // A fragment base: lane holds row (rowBase + tm*16 + lm),
    // elements j=0..15 -> K = k0 + kg*16 + j.
    const unsigned short* aRow0 = A + (size_t)(rowBase + lm) * K + kg * 16;
    const unsigned short* aRow1 = aRow0 + (size_t)16 * K;

    // B fragment base: lane holds col (colBase + tn*16 + lm),
    // elements j=0..15 -> K = k0 + kg*16 + j (contiguous 32B in Bt).
    const unsigned short* bCol[4];
#pragma unroll
    for (int tn = 0; tn < 4; ++tn)
        bCol[tn] = Bt + (size_t)(colBase + tn * 16 + lm) * K + kg * 16;

#pragma unroll 2
    for (int k0 = 0; k0 < K; k0 += 32) {
        v16bf afrag[2];
        {
            v8bf lo = *(const v8bf*)(aRow0 + k0);
            v8bf hi = *(const v8bf*)(aRow0 + k0 + 8);
            afrag[0] = __builtin_shufflevector(lo, hi,
                0,1,2,3,4,5,6,7,8,9,10,11,12,13,14,15);
        }
        {
            v8bf lo = *(const v8bf*)(aRow1 + k0);
            v8bf hi = *(const v8bf*)(aRow1 + k0 + 8);
            afrag[1] = __builtin_shufflevector(lo, hi,
                0,1,2,3,4,5,6,7,8,9,10,11,12,13,14,15);
        }
        v16bf bfrag[4];
#pragma unroll
        for (int tn = 0; tn < 4; ++tn)
            bfrag[tn] = *(const v16bf*)(bCol[tn] + k0);

#pragma unroll
        for (int tm = 0; tm < 2; ++tm)
#pragma unroll
            for (int tn = 0; tn < 4; ++tn)
                acc[tm][tn] = __builtin_amdgcn_wmma_f32_16x16x32_bf16(
                    /*neg_a=*/false, afrag[tm],
                    /*neg_b=*/false, bfrag[tn],
                    /*c_mod=*/(short)0, acc[tm][tn],
                    /*reuse_a=*/false, /*reuse_b=*/false);
    }

    // Epilogue: C/D layout -> VGPR r: lanes 0-15 hold M=r, lanes 16-31 M=8+r.
#pragma unroll
    for (int tm = 0; tm < 2; ++tm) {
#pragma unroll
        for (int tn = 0; tn < 4; ++tn) {
            int m0 = rowBase + tm * 16 + kg * 8;
            int n  = colBase + tn * 16 + lm;
            float bv = bias[n];
            float* cp = C + (size_t)m0 * N + n;
#pragma unroll
            for (int r = 0; r < 8; ++r)
                cp[(size_t)r * N] = acc[tm][tn][r] + bv;
        }
    }
}

// -------------------------- fused LayerNorm + LIF --------------------------
// One block per (b,l). Rows t*BL + bl for t=0..3 share LIF state per column.
// COLS = D/256 columns per thread, held in registers across timesteps.

template <int COLS, bool OUT_BF16>
__global__ __launch_bounds__(256) void lnlif_kernel(
    const float* __restrict__ X,        // [4*BL, D] pre-activations
    const float* __restrict__ gamma,    // [D]
    const float* __restrict__ beta,     // [D]
    void* __restrict__ out) {           // spikes: bf16 or fp32, same layout

    constexpr int D = COLS * 256;
    constexpr float INV_TAU = 1.0f / 1.2f;
    __shared__ float sSum[256];
    __shared__ float sSq[256];

    const int tid = threadIdx.x;
    const int bl  = blockIdx.x;

    float v[COLS], g[COLS], be[COLS];
#pragma unroll
    for (int c = 0; c < COLS; ++c) {
        v[c]  = 0.0f;
        g[c]  = gamma[tid + 256 * c];
        be[c] = beta[tid + 256 * c];
    }

    for (int t = 0; t < T_STEPS; ++t) {
        size_t base = (size_t)(t * BL + bl) * D;
        float xv[COLS];
        float s = 0.0f, q = 0.0f;
#pragma unroll
        for (int c = 0; c < COLS; ++c) {
            xv[c] = X[base + tid + 256 * c];
            s += xv[c];
            q += xv[c] * xv[c];
        }
        __syncthreads();               // protect shared reuse across timesteps
        sSum[tid] = s;
        sSq[tid]  = q;
        __syncthreads();
#pragma unroll
        for (int off = 128; off > 0; off >>= 1) {
            if (tid < off) {
                sSum[tid] += sSum[tid + off];
                sSq[tid]  += sSq[tid + off];
            }
            __syncthreads();
        }
        float mean = sSum[0] * (1.0f / D);
        float var  = sSq[0] * (1.0f / D) - mean * mean;
        float rs   = rsqrtf(var + 1e-12f);

#pragma unroll
        for (int c = 0; c < COLS; ++c) {
            float y  = (xv[c] - mean) * rs * g[c] + be[c];
            float vv = v[c] + (y - v[c]) * INV_TAU;   // LIF charge
            float spk = (vv >= 1.0f) ? 1.0f : 0.0f;   // threshold 1.0
            v[c] = (vv >= 1.0f) ? 0.0f : vv;          // hard reset
            size_t oi = base + tid + 256 * c;
            if (OUT_BF16) ((unsigned short*)out)[oi] = f2bf(spk);
            else          ((float*)out)[oi] = spk;
        }
    }
}

// ------------------------------- launcher ----------------------------------

extern "C" void kernel_launch(void* const* d_in, const int* in_sizes, int n_in,
                              void* d_out, int out_size, void* d_ws, size_t ws_size,
                              hipStream_t stream) {
    (void)in_sizes; (void)n_in; (void)out_size; (void)ws_size;

    const float* x   = (const float*)d_in[0];
    const float* W1  = (const float*)d_in[1];
    const float* b1  = (const float*)d_in[2];
    const float* g1  = (const float*)d_in[3];
    const float* be1 = (const float*)d_in[4];
    const float* W2  = (const float*)d_in[5];
    const float* b2  = (const float*)d_in[6];
    const float* g2  = (const float*)d_in[7];
    const float* be2 = (const float*)d_in[8];

    // Workspace carve-up (256B aligned)
    char* ws = (char*)d_ws;
    size_t off = 0;
    auto carve = [&](size_t bytes) {
        size_t o = off;
        off = (off + bytes + 255) & ~(size_t)255;
        return o;
    };
    unsigned short* Xb  = (unsigned short*)(ws + carve((size_t)M_ROWS * D_DIM * 2));
    unsigned short* W1t = (unsigned short*)(ws + carve((size_t)H_DIM  * D_DIM * 2));
    unsigned short* W2t = (unsigned short*)(ws + carve((size_t)D_DIM  * H_DIM * 2));
    float*          C1  = (float*)         (ws + carve((size_t)M_ROWS * H_DIM * 4));
    unsigned short* S1  = (unsigned short*)(ws + carve((size_t)M_ROWS * H_DIM * 2));
    float*          C2  = (float*)         (ws + carve((size_t)M_ROWS * D_DIM * 4));

    // 1) x fp32 -> bf16
    {
        int n = M_ROWS * D_DIM;
        f32_to_bf16_kernel<<<(n + 255) / 256, 256, 0, stream>>>(x, Xb, n);
    }
    // 2) W1 [768,3072] -> W1t [3072,768] bf16
    transpose_to_bf16_kernel<<<dim3(H_DIM / 256, D_DIM), 256, 0, stream>>>(
        W1, W1t, D_DIM, H_DIM);
    // 3) W2 [3072,768] -> W2t [768,3072] bf16
    transpose_to_bf16_kernel<<<dim3(D_DIM / 256, H_DIM), 256, 0, stream>>>(
        W2, W2t, H_DIM, D_DIM);

    // 4) fc1: C1[M,H] = Xb @ W1 + b1
    wmma_gemm_bf16_kernel<<<dim3(H_DIM / 128, M_ROWS / 128), 256, 0, stream>>>(
        Xb, W1t, b1, C1, M_ROWS, H_DIM, D_DIM);

    // 5) LN1 + LIF1 -> bf16 spikes S1
    lnlif_kernel<H_DIM / 256, true><<<BL, 256, 0, stream>>>(C1, g1, be1, (void*)S1);

    // 6) fc2: C2[M,D] = S1 @ W2 + b2
    wmma_gemm_bf16_kernel<<<dim3(D_DIM / 128, M_ROWS / 128), 256, 0, stream>>>(
        S1, W2t, b2, C2, M_ROWS, D_DIM, H_DIM);

    // 7) LN2 + LIF2 -> fp32 spikes to d_out
    lnlif_kernel<D_DIM / 256, false><<<BL, 256, 0, stream>>>(
        C2, g2, be2, d_out);
}
